// MaskDINOEncoderHeadModule_22436909154519
// MI455X (gfx1250) — compile-verified, matrix operand order checked by reading, not run on hardware
//
#include <hip/hip_runtime.h>

// ---------------------------------------------------------------------------
// MaskDINO encoder for MI455X (gfx1250, wave32).
// GEMMs run on V_WMMA_F32_16X16X32_F16 (f16 inputs, f32 accumulate).
// ---------------------------------------------------------------------------

typedef __attribute__((ext_vector_type(16))) _Float16 v16h;
typedef __attribute__((ext_vector_type(8)))  _Float16 v8h;
typedef __attribute__((ext_vector_type(8)))  float    v8f;

#define NH_  8
#define NL_  4
#define NP_  4
#define D_   256
#define DFF_ 1024
#define LQ_  21760   // 128*128 + 64*64 + 32*32 + 16*16

// ---------------------------------------------------------------------------
// Tiled WMMA GEMM: C[M,N] = A[M,K](f16) * B[K,N](f16) + bias, opt ReLU,
// output either f32 or f16. Block = 256 thr = 8 waves; tile 128(M) x 128(N),
// K-step 32. Waves tiled 4(M) x 2(N); each wave owns 32x64 = 2x4 WMMA frags,
// so 8 WMMAs per K-step are fed by only 12 ds_load_b128 (A frags reused 4x,
// B frags 2x) -- DS bandwidth per FLOP halved vs a 32x32 wave tile.
// ---------------------------------------------------------------------------
#define TM 128
#define TN 128
#define TK 32

template <bool RELU, bool OUT_HALF>
__global__ __launch_bounds__(256)
void gemm_wmma_f16(const _Float16* __restrict__ A,   // [M,K]
                   const _Float16* __restrict__ Bw,  // [K,N]
                   const float*    __restrict__ bias,// [N]
                   float*          Cf,               // [M,N] (if !OUT_HALF)
                   _Float16*       Ch,               // [M,N] (if OUT_HALF)
                   int M, int N, int K) {
  __shared__ __align__(16) _Float16 As[TM * TK];   // [m][k]
  __shared__ __align__(16) _Float16 Bs[TN * TK];   // [n][k] (transposed)

  const int tid  = threadIdx.x;
  const int m0   = blockIdx.y * TM;
  const int n0   = blockIdx.x * TN;
  const int wave = tid >> 5;
  const int lane = tid & 31;
  const int wm   = (wave & 3) * 32;   // wave M offset in tile (4 waves down M)
  const int wn   = (wave >> 2) * 64;  // wave N offset in tile (2 waves across N)
  const int l16  = lane & 15;
  const int lh   = lane >> 4;

  v8f c[2][4] = {};

  const int ar    = tid >> 1;           // A row this thread stages (0..127)
  const int apart = (tid & 1) * 16;     // first/second 16 halves of the row
  const int bk    = tid >> 3;           // B k-row this thread stages (0..31)
  const int bn    = (tid & 7) * 16;     // 16 consecutive n

  for (int k0 = 0; k0 < K; k0 += TK) {
    // ---- stage A tile (128x32 halves), already f16 in global ----
    {
      const _Float16* gp = A + (size_t)(m0 + ar) * K + k0 + apart;
      v8h x0 = *(const v8h*)gp;
      v8h x1 = *(const v8h*)(gp + 8);
      *(v8h*)&As[ar * TK + apart]     = x0;
      *(v8h*)&As[ar * TK + apart + 8] = x1;
      if (k0 + TK < K)  // prefetch next K-slab (global_prefetch_b8)
        __builtin_prefetch(gp + TK, 0, 1);
    }
    // ---- stage B tile (32x128 halves), transpose into Bs[n][k] ----
    {
      const _Float16* gp = Bw + (size_t)(k0 + bk) * N + n0 + bn;
      v8h x0 = *(const v8h*)gp;
      v8h x1 = *(const v8h*)(gp + 8);
#pragma unroll
      for (int j = 0; j < 8; ++j) Bs[(bn + j) * TK + bk] = x0[j];
#pragma unroll
      for (int j = 0; j < 8; ++j) Bs[(bn + 8 + j) * TK + bk] = x1[j];
    }
    __syncthreads();

    // ---- fragments. ISA 16-bit A layout: lane m = lane%16; halves 0-7
    // hold K = h*8+j, halves 8-15 hold K = 16+h*8+j  (h = lane/16).
    // B mirrors this with lane = column n. ----
    v16h af[2], bf[4];
#pragma unroll
    for (int mi = 0; mi < 2; ++mi) {
      const _Float16* p = &As[(wm + mi * 16 + l16) * TK];
      v8h lo = *(const v8h*)(p + lh * 8);
      v8h hi = *(const v8h*)(p + 16 + lh * 8);
      af[mi] = __builtin_shufflevector(lo, hi, 0,1,2,3,4,5,6,7,8,9,10,11,12,13,14,15);
    }
#pragma unroll
    for (int ni = 0; ni < 4; ++ni) {
      const _Float16* p = &Bs[(wn + ni * 16 + l16) * TK];
      v8h lo = *(const v8h*)(p + lh * 8);
      v8h hi = *(const v8h*)(p + 16 + lh * 8);
      bf[ni] = __builtin_shufflevector(lo, hi, 0,1,2,3,4,5,6,7,8,9,10,11,12,13,14,15);
    }
#pragma unroll
    for (int mi = 0; mi < 2; ++mi)
#pragma unroll
      for (int ni = 0; ni < 4; ++ni)
        c[mi][ni] = __builtin_amdgcn_wmma_f32_16x16x32_f16(
            false, af[mi], false, bf[ni], (short)0, c[mi][ni], false, false);
    __syncthreads();
  }

  // ---- epilogue. C layout: VGPR r holds (M = h*8+r, N = lane%16). ----
#pragma unroll
  for (int mi = 0; mi < 2; ++mi)
#pragma unroll
    for (int ni = 0; ni < 4; ++ni) {
      const int col = n0 + wn + ni * 16 + l16;
      const float bv = bias[col];
#pragma unroll
      for (int r = 0; r < 8; ++r) {
        const int row = m0 + wm + mi * 16 + lh * 8 + r;
        float v = c[mi][ni][r] + bv;
        if (RELU) v = v > 0.f ? v : 0.f;
        if (OUT_HALF) Ch[(size_t)row * N + col] = (_Float16)v;
        else          Cf[(size_t)row * N + col] = v;
      }
    }
}

// ---------------------------------------------------------------------------
// Channel-first [B,D,H,W] inputs -> token-major [B*Lq, D]; pos gets level emb.
// ---------------------------------------------------------------------------
__global__ __launch_bounds__(256)
void prep_tokens(const float* __restrict__ src_l, const float* __restrict__ pos_l,
                 const float* __restrict__ lvl_emb, float* __restrict__ src_tok,
                 float* __restrict__ pos_tok, int HW, int start, int Lq) {
  const int bt = blockIdx.x;
  const int b  = bt / HW, t = bt - b * HW;
  const int d  = threadIdx.x;
  const size_t gi = ((size_t)b * Lq + start + t) * D_ + d;
  const size_t si = ((size_t)b * D_ + d) * HW + t;
  src_tok[gi] = src_l[si];
  pos_tok[gi] = pos_l[si] + lvl_emb[d];
}

// ---------------------------------------------------------------------------
// Elementwise helpers
// ---------------------------------------------------------------------------
__global__ void f32_to_f16_k(const float* __restrict__ a, _Float16* __restrict__ o, size_t n) {
  size_t i = (size_t)blockIdx.x * blockDim.x + threadIdx.x;
  if (i < n) o[i] = (_Float16)a[i];
}
__global__ void add_to_f16_k(const float* __restrict__ a, const float* __restrict__ b,
                             _Float16* __restrict__ o, size_t n) {
  size_t i = (size_t)blockIdx.x * blockDim.x + threadIdx.x;
  if (i < n) o[i] = (_Float16)(a[i] + b[i]);
}
__global__ void copy_f32_k(const float* __restrict__ a, float* __restrict__ o, size_t n) {
  size_t i = (size_t)blockIdx.x * blockDim.x + threadIdx.x;
  if (i < n) o[i] = a[i];
}

// ---------------------------------------------------------------------------
// Softmax over groups of 16 (NL*NP per head); aw is [M, NH*16] contiguous.
// ---------------------------------------------------------------------------
__global__ void softmax16_k(float* __restrict__ aw, size_t ngroups) {
  size_t i = (size_t)blockIdx.x * blockDim.x + threadIdx.x;
  if (i >= ngroups) return;
  float* p = aw + i * 16;
  float mx = p[0];
#pragma unroll
  for (int j = 1; j < 16; ++j) mx = fmaxf(mx, p[j]);
  float s = 0.f;
#pragma unroll
  for (int j = 0; j < 16; ++j) { float e = __expf(p[j] - mx); p[j] = e; s += e; }
  const float inv = 1.f / s;
#pragma unroll
  for (int j = 0; j < 16; ++j) p[j] *= inv;
}

// ---------------------------------------------------------------------------
// Multi-scale deformable attention sampling.
// grid = B*Lq blocks; 8 waves = 8 heads; lane = channel (DH=32) so each
// corner fetch is one contiguous 128B line (L2-resident `val`).
// ---------------------------------------------------------------------------
__global__ __launch_bounds__(256)
void deform_attn_k(const float* __restrict__ val,   // [B*Lq, NH*32]
                   const float* __restrict__ off,   // [B*Lq, NH*NL*NP*2]
                   const float* __restrict__ aw,    // [B*Lq, NH*NL*NP]
                   _Float16* __restrict__ attn_h,   // [B*Lq, 256]
                   int Lq) {
  const int qg   = blockIdx.x;
  const int h    = threadIdx.x >> 5;
  const int lane = threadIdx.x & 31;
  const int b    = qg / Lq;
  const int qi   = qg - b * Lq;

  const int LH[4] = {128, 64, 32, 16};
  const int LW[4] = {128, 64, 32, 16};
  const int LS[4] = {0, 16384, 20480, 21504};

  int l0 = 0;
  if (qi >= 21504) l0 = 3; else if (qi >= 20480) l0 = 2; else if (qi >= 16384) l0 = 1;
  const int t   = qi - LS[l0];
  const int Wl0 = LW[l0];
  const float refx = ((t % Wl0) + 0.5f) / (float)Wl0;
  const float refy = ((t / Wl0) + 0.5f) / (float)LH[l0];

  const float* offq = off + (size_t)qg * (NH_ * NL_ * NP_ * 2) + h * (NL_ * NP_ * 2);
  const float* awq  = aw  + (size_t)qg * (NH_ * NL_ * NP_)     + h * (NL_ * NP_);

  float acc = 0.f;
  for (int l = 0; l < NL_; ++l) {
    const int Hh = LH[l], Ww = LW[l];
    const size_t base = ((size_t)b * Lq + LS[l]) * D_ + h * 32 + lane;
#pragma unroll
    for (int p = 0; p < NP_; ++p) {
      const float ox = offq[(l * NP_ + p) * 2 + 0];
      const float oy = offq[(l * NP_ + p) * 2 + 1];
      const float w  = awq[l * NP_ + p];
      const float x  = (refx + ox / (float)Ww) * Ww - 0.5f;
      const float y  = (refy + oy / (float)Hh) * Hh - 0.5f;
      const float xf = floorf(x), yf = floorf(y);
      const int   x0 = (int)xf, y0 = (int)yf;
      const float wx = x - xf,  wy = y - yf;
      float s = 0.f;
#pragma unroll
      for (int cy = 0; cy < 2; ++cy)
#pragma unroll
        for (int cx = 0; cx < 2; ++cx) {
          const int ix = x0 + cx, iy = y0 + cy;
          if (ix >= 0 && ix < Ww && iy >= 0 && iy < Hh) {
            const float cw = (cx ? wx : 1.f - wx) * (cy ? wy : 1.f - wy);
            s += cw * val[base + (size_t)(iy * Ww + ix) * D_];
          }
        }
      acc += w * s;
    }
  }
  attn_h[(size_t)qg * D_ + h * 32 + lane] = (_Float16)acc;
}

// ---------------------------------------------------------------------------
// Fused residual-add + LayerNorm (D=256, one block per token), optional f16
// copy of the normalized output for the next GEMM. In-place safe (per-row).
// ---------------------------------------------------------------------------
__global__ __launch_bounds__(256)
void add_layernorm_k(const float* x, const float* dlt, const float* g,
                     const float* bb, float* out_f, _Float16* out_h) {
  __shared__ float sbuf[256];
  const int m = blockIdx.x, d = threadIdx.x;
  const size_t idx = (size_t)m * D_ + d;
  const float v = x[idx] + dlt[idx];
  sbuf[d] = v;
  __syncthreads();
  for (int s = 128; s > 0; s >>= 1) { if (d < s) sbuf[d] += sbuf[d + s]; __syncthreads(); }
  const float mu = sbuf[0] * (1.f / 256.f);
  __syncthreads();
  const float dv = v - mu;
  sbuf[d] = dv * dv;
  __syncthreads();
  for (int s = 128; s > 0; s >>= 1) { if (d < s) sbuf[d] += sbuf[d + s]; __syncthreads(); }
  const float var = sbuf[0] * (1.f / 256.f);
  const float y = dv * rsqrtf(var + 1e-5f) * g[d] + bb[d];
  out_f[idx] = y;
  if (out_h) out_h[idx] = (_Float16)y;
}

// ---------------------------------------------------------------------------
// Host-side orchestration
// ---------------------------------------------------------------------------
extern "C" void kernel_launch(void* const* d_in, const int* in_sizes, int n_in,
                              void* d_out, int out_size, void* d_ws, size_t ws_size,
                              hipStream_t stream) {
  const float* src_in[4] = {(const float*)d_in[0], (const float*)d_in[2],
                            (const float*)d_in[4], (const float*)d_in[6]};
  const float* pos_in[4] = {(const float*)d_in[1], (const float*)d_in[3],
                            (const float*)d_in[5], (const float*)d_in[7]};
  const float* level_embed = (const float*)d_in[8];
  const float* W_off = (const float*)d_in[9];   const float* b_off = (const float*)d_in[10];
  const float* W_aw  = (const float*)d_in[11];  const float* b_aw  = (const float*)d_in[12];
  const float* W_val = (const float*)d_in[13];  const float* b_val = (const float*)d_in[14];
  const float* W_out = (const float*)d_in[15];  const float* b_out = (const float*)d_in[16];
  const float* ln1_g = (const float*)d_in[17];  const float* ln1_b = (const float*)d_in[18];
  const float* W_ff1 = (const float*)d_in[19];  const float* b_ff1 = (const float*)d_in[20];
  const float* W_ff2 = (const float*)d_in[21];  const float* b_ff2 = (const float*)d_in[22];
  const float* ln2_g = (const float*)d_in[23];  const float* ln2_b = (const float*)d_in[24];

  const int B  = in_sizes[0] / (D_ * 128 * 128);
  const int Lq = LQ_;
  const int M  = B * Lq;  // 43520 for B=2; always divisible by 128 (21760/128=170)

  // ---- workspace carve (256B aligned) ----
  char* ws = (char*)d_ws;
  auto carve = [&](size_t bytes) {
    void* p = (void*)ws;
    ws += (bytes + 255) & ~(size_t)255;
    return p;
  };
  float*    pos_tok = (float*)   carve((size_t)M * D_ * 4);
  float*    src_tok = (float*)   carve((size_t)M * D_ * 4);
  _Float16* q_h     = (_Float16*)carve((size_t)M * D_ * 2);
  _Float16* src_h   = (_Float16*)carve((size_t)M * D_ * 2);
  float*    off_f   = (float*)   carve((size_t)M * D_ * 4);
  float*    aw_f    = (float*)   carve((size_t)M * 128 * 4);
  float*    val_f   = (float*)   carve((size_t)M * D_ * 4);
  _Float16* attn_h  = (_Float16*)carve((size_t)M * D_ * 2);
  float*    delta_f = (float*)   carve((size_t)M * D_ * 4);
  _Float16* ff_h    = (_Float16*)carve((size_t)M * DFF_ * 2);
  _Float16* Wh_off  = (_Float16*)carve((size_t)6 * D_ * D_ * 2);
  _Float16* Wh_aw   = (_Float16*)carve((size_t)6 * D_ * 128 * 2);
  _Float16* Wh_val  = (_Float16*)carve((size_t)6 * D_ * D_ * 2);
  _Float16* Wh_out  = (_Float16*)carve((size_t)6 * D_ * D_ * 2);
  _Float16* Wh_ff1  = (_Float16*)carve((size_t)6 * D_ * DFF_ * 2);
  _Float16* Wh_ff2  = (_Float16*)carve((size_t)6 * DFF_ * D_ * 2);
  (void)ws_size; (void)n_in; (void)out_size;

  auto ew_grid = [](size_t n) { return (unsigned)((n + 255) / 256); };

  // ---- one-time (per launch) weight conversion to f16 ----
  f32_to_f16_k<<<ew_grid(6 * D_ * D_), 256, 0, stream>>>(W_off, Wh_off, (size_t)6 * D_ * D_);
  f32_to_f16_k<<<ew_grid(6 * D_ * 128), 256, 0, stream>>>(W_aw, Wh_aw, (size_t)6 * D_ * 128);
  f32_to_f16_k<<<ew_grid(6 * D_ * D_), 256, 0, stream>>>(W_val, Wh_val, (size_t)6 * D_ * D_);
  f32_to_f16_k<<<ew_grid(6 * D_ * D_), 256, 0, stream>>>(W_out, Wh_out, (size_t)6 * D_ * D_);
  f32_to_f16_k<<<ew_grid(6 * D_ * DFF_), 256, 0, stream>>>(W_ff1, Wh_ff1, (size_t)6 * D_ * DFF_);
  f32_to_f16_k<<<ew_grid(6 * DFF_ * D_), 256, 0, stream>>>(W_ff2, Wh_ff2, (size_t)6 * DFF_ * D_);

  // ---- flatten channel-first inputs to token-major; add level embedding ----
  const int lvlHW[4]    = {16384, 4096, 1024, 256};
  const int lvlStart[4] = {0, 16384, 20480, 21504};
  for (int l = 0; l < 4; ++l)
    prep_tokens<<<B * lvlHW[l], 256, 0, stream>>>(src_in[l], pos_in[l],
        level_embed + l * D_, src_tok, pos_tok, lvlHW[l], lvlStart[l], Lq);

  const dim3 blk(256);
  const dim3 gD(D_ / TN, M / TM);     // N=256 -> x=2
  const dim3 gA(128 / TN, M / TM);    // N=128 -> x=1
  const dim3 gF(DFF_ / TN, M / TM);   // N=1024 -> x=8
  const size_t nMD = (size_t)M * D_;

  for (int i = 0; i < 6; ++i) {
    // q = src + pos  (f16 for WMMA)
    add_to_f16_k<<<ew_grid(nMD), blk, 0, stream>>>(src_tok, pos_tok, q_h, nMD);
    // sampling offsets and attention logits
    gemm_wmma_f16<false, false><<<gD, blk, 0, stream>>>(
        q_h, Wh_off + (size_t)i * D_ * D_, b_off + i * D_, off_f, nullptr, M, D_, D_);
    gemm_wmma_f16<false, false><<<gA, blk, 0, stream>>>(
        q_h, Wh_aw + (size_t)i * D_ * 128, b_aw + i * 128, aw_f, nullptr, M, 128, D_);
    softmax16_k<<<ew_grid((size_t)M * NH_), blk, 0, stream>>>(aw_f, (size_t)M * NH_);
    // value projection (input is src, not q)
    f32_to_f16_k<<<ew_grid(nMD), blk, 0, stream>>>(src_tok, src_h, nMD);
    gemm_wmma_f16<false, false><<<gD, blk, 0, stream>>>(
        src_h, Wh_val + (size_t)i * D_ * D_, b_val + i * D_, val_f, nullptr, M, D_, D_);
    // deformable sampling -> f16 attn
    deform_attn_k<<<M, blk, 0, stream>>>(val_f, off_f, aw_f, attn_h, Lq);
    // output projection
    gemm_wmma_f16<false, false><<<gD, blk, 0, stream>>>(
        attn_h, Wh_out + (size_t)i * D_ * D_, b_out + i * D_, delta_f, nullptr, M, D_, D_);
    // src = LN(src + attn); also emit f16 for FFN
    add_layernorm_k<<<M, blk, 0, stream>>>(src_tok, delta_f,
        ln1_g + i * D_, ln1_b + i * D_, src_tok, src_h);
    // FFN: ReLU fused, f16 output feeds second GEMM directly
    gemm_wmma_f16<true, true><<<gF, blk, 0, stream>>>(
        src_h, Wh_ff1 + (size_t)i * D_ * DFF_, b_ff1 + i * DFF_, nullptr, ff_h, M, DFF_, D_);
    gemm_wmma_f16<false, false><<<gD, blk, 0, stream>>>(
        ff_h, Wh_ff2 + (size_t)i * DFF_ * D_, b_ff2 + i * D_, delta_f, nullptr, M, D_, DFF_);
    // src = LN(src + ff)
    add_layernorm_k<<<M, blk, 0, stream>>>(src_tok, delta_f,
        ln2_g + i * D_, ln2_b + i * D_, src_tok, nullptr);
  }

  copy_f32_k<<<ew_grid(nMD), blk, 0, stream>>>(src_tok, (float*)d_out, nMD);
}
